// SampledSoftmaxLoss_66314295050861
// MI455X (gfx1250) — compile-verified, matrix operand order checked by reading, not run on hardware
//
#include <hip/hip_runtime.h>
#include <hip/hip_bf16.h>

typedef __attribute__((ext_vector_type(16))) _Float16 v16h;
typedef __attribute__((ext_vector_type(8)))  _Float16 v8h;
typedef __attribute__((ext_vector_type(4)))  _Float16 v4h;
typedef __attribute__((ext_vector_type(8)))  float    v8f;

namespace {
constexpr int   kVocab   = 100000;
constexpr int   kHidden  = 128;
constexpr int   kTok     = 1024;
constexpr int   kSamp    = 4096;
constexpr int   kGroups  = kSamp / 16;   // 256 tiles of 16 samples per token
constexpr int   kWaves   = 8;            // waves per block (256 threads, wave32)
constexpr int   kIgnore  = -100;
constexpr float kEps     = 1e-6f;

// workspace layout (float indices / byte offsets)
constexpr int    kInvnOff   = 0;                           // 100000 f32
constexpr int    kXnormOff  = 100352;                      // 1024*128 f32
constexpr int    kAccumOff  = kXnormOff + kTok * kHidden;  // 2 f32
constexpr size_t kProjhByte = 925952;                      // 256B aligned
constexpr size_t kProjhSize = (size_t)kVocab * kHidden * sizeof(_Float16); // 25.6 MB
constexpr size_t kWsNeeded  = kProjhByte + kProjhSize;
}

__device__ __forceinline__ float wave_sum32(float v) {
#pragma unroll
  for (int off = 16; off >= 1; off >>= 1) v += __shfl_xor(v, off, 32);
  return v;
}

// ------- kernel 1: inverse norms + (optionally) pre-normalized f16 table -------
__global__ __launch_bounds__(256) void proj_prep_kernel(
    const float* __restrict__ proj, float* __restrict__ invn,
    _Float16* __restrict__ projh, int nrows, int write_h) {
  const int lane = threadIdx.x & 31;
  const int wid  = (blockIdx.x * blockDim.x + threadIdx.x) >> 5;
  const int nw   = (gridDim.x * blockDim.x) >> 5;
  for (int row = wid; row < nrows; row += nw) {
    const float4 v = ((const float4*)(proj + (long)row * kHidden))[lane];
    float ss = v.x * v.x + v.y * v.y + v.z * v.z + v.w * v.w;
    ss = wave_sum32(ss);                                 // all lanes hold total
    const float inv = 1.0f / fmaxf(sqrtf(ss), kEps);
    if (lane == 0) invn[row] = inv;
    if (write_h) {
      v4h h;
      h[0] = (_Float16)(v.x * inv); h[1] = (_Float16)(v.y * inv);
      h[2] = (_Float16)(v.z * inv); h[3] = (_Float16)(v.w * inv);
      *(v4h*)(projh + (long)row * kHidden + lane * 4) = h;  // 8B store
    }
  }
}

// ---------------- kernel 2: normalize inputs, zero accumulators ----------------
__global__ __launch_bounds__(256) void x_normalize_kernel(
    const float* __restrict__ x, float* __restrict__ xn, float* __restrict__ accum) {
  if (blockIdx.x == 0 && threadIdx.x == 0) { accum[0] = 0.0f; accum[1] = 0.0f; }
  const int lane = threadIdx.x & 31;
  const int wid  = (blockIdx.x * blockDim.x + threadIdx.x) >> 5;
  const int nw   = (gridDim.x * blockDim.x) >> 5;
  for (int row = wid; row < kTok; row += nw) {
    const float4 v = ((const float4*)(x + (long)row * kHidden))[lane];
    float ss = v.x * v.x + v.y * v.y + v.z * v.z + v.w * v.w;
    ss = wave_sum32(ss);
    const float inv = 1.0f / fmaxf(sqrtf(ss), kEps);
    float4 o; o.x = v.x * inv; o.y = v.y * inv; o.z = v.z * inv; o.w = v.w * inv;
    ((float4*)(xn + (long)row * kHidden))[lane] = o;
  }
}

// ---------------- kernel 3: per-token sampled softmax via WMMA ----------------
// USE_F16: gather pre-normalized f16 rows (256B/row, no in-loop scale/convert).
// Fallback: gather f32 rows and scale by invnorm at load (512B/row).
template <bool USE_F16>
__global__ __launch_bounds__(256, 2) void sampled_nce_main_kernel(
    const float* __restrict__ proj, const _Float16* __restrict__ projh,
    const int* __restrict__ labels, const int* __restrict__ samples,
    const float* __restrict__ invnorm, const float* __restrict__ xnorm,
    float* __restrict__ accum) {
  __shared__ float s_x[kHidden];
  __shared__ float s_m[kWaves], s_s[kWaves];

  const int tok  = blockIdx.x;
  const int tid  = threadIdx.x;
  const int lane = tid & 31;
  const int wave = tid >> 5;

  if (tid < kHidden) s_x[tid] = xnorm[(long)tok * kHidden + tid];
  __syncthreads();

  // Loop-invariant B operand: x replicated into all 16 columns.
  // Per-lane K mapping mirrors the 16-bit A layout: lanes 0-15 -> K {0..7,16..23},
  // lanes 16-31 -> K {8..15,24..31} within each 32-wide chunk.
  const int b0 = (lane < 16) ? 0 : 8;
  v16h B[4];
#pragma unroll
  for (int c = 0; c < 4; ++c) {
#pragma unroll
    for (int i = 0; i < 8; ++i) {
      B[c][i]     = (_Float16)s_x[c * 32 + b0 + i];
      B[c][i + 8] = (_Float16)s_x[c * 32 + b0 + 16 + i];
    }
  }

  const int  label = labels[tok];
  const int* srow  = samples + (long)tok * kSamp + (lane & 15);
  const int  r0    = (lane < 16) ? 0 : 8;

  float m = -1e30f, s = 0.0f;   // per-lane online logsumexp state

  int sidx = srow[wave * 16];   // first tile's row index for this lane

  for (int g = wave; g < kGroups; g += kWaves) {
    // fetch next tile's index once; reuse for prefetch and next iteration
    const int gn = g + kWaves;
    int next_sidx = 0;
    if (gn < kGroups) {
      next_sidx = srow[gn * 16];
      if (USE_F16) __builtin_prefetch(projh + (long)next_sidx * kHidden, 0, 1);
      else         __builtin_prefetch(proj  + (long)next_sidx * kHidden, 0, 1);
    }

    // ---- load all four A fragments (independent registers), then 4
    // ---- *independent* WMMAs (separate accumulators -> no serial C-chain,
    // ---- scheduler can keep all 8 b128 loads in flight with partial waits)
    v16h a[4];
    if constexpr (USE_F16) {
      const _Float16* pr = projh + (long)sidx * kHidden;
#pragma unroll
      for (int c = 0; c < 4; ++c) {
        const v8h lo = *(const v8h*)(pr + c * 32 + b0);       // 16B b128 load
        const v8h hi = *(const v8h*)(pr + c * 32 + b0 + 16);  // 16B b128 load
#pragma unroll
        for (int i = 0; i < 8; ++i) { a[c][i] = lo[i]; a[c][i + 8] = hi[i]; }
      }
    } else {
      const float  invn = invnorm[sidx];
      const float* prow = proj + (long)sidx * kHidden;
#pragma unroll
      for (int c = 0; c < 4; ++c) {
        const float4 lo0 = ((const float4*)(prow + c * 32 + b0))[0];
        const float4 lo1 = ((const float4*)(prow + c * 32 + b0))[1];
        const float4 hi0 = ((const float4*)(prow + c * 32 + b0 + 16))[0];
        const float4 hi1 = ((const float4*)(prow + c * 32 + b0 + 16))[1];
        a[c][0]  = (_Float16)(lo0.x * invn); a[c][1]  = (_Float16)(lo0.y * invn);
        a[c][2]  = (_Float16)(lo0.z * invn); a[c][3]  = (_Float16)(lo0.w * invn);
        a[c][4]  = (_Float16)(lo1.x * invn); a[c][5]  = (_Float16)(lo1.y * invn);
        a[c][6]  = (_Float16)(lo1.z * invn); a[c][7]  = (_Float16)(lo1.w * invn);
        a[c][8]  = (_Float16)(hi0.x * invn); a[c][9]  = (_Float16)(hi0.y * invn);
        a[c][10] = (_Float16)(hi0.z * invn); a[c][11] = (_Float16)(hi0.w * invn);
        a[c][12] = (_Float16)(hi1.x * invn); a[c][13] = (_Float16)(hi1.y * invn);
        a[c][14] = (_Float16)(hi1.z * invn); a[c][15] = (_Float16)(hi1.w * invn);
      }
    }

    const v8f zero = {0.f, 0.f, 0.f, 0.f, 0.f, 0.f, 0.f, 0.f};
    v8f acc0 = __builtin_amdgcn_wmma_f32_16x16x32_f16(
        false, a[0], false, B[0], (short)0, zero, false, false);
    v8f acc1 = __builtin_amdgcn_wmma_f32_16x16x32_f16(
        false, a[1], false, B[1], (short)0, zero, false, false);
    v8f acc2 = __builtin_amdgcn_wmma_f32_16x16x32_f16(
        false, a[2], false, B[2], (short)0, zero, false, false);
    v8f acc3 = __builtin_amdgcn_wmma_f32_16x16x32_f16(
        false, a[3], false, B[3], (short)0, zero, false, false);
    const v8f acc = (acc0 + acc1) + (acc2 + acc3);   // v_pk_add_f32 merge

    // lane holds dot results for rows r0..r0+7 (replicated across 16 lanes)
#pragma unroll
    for (int k = 0; k < 8; ++k) {
      float val = acc[k];
      const int rsidx = __shfl(sidx, r0 + k, 32);
      val = (rsidx == label) ? (val - 1e6f) : val;
      const float mn = fmaxf(m, val);
      s = s * __expf(m - mn) + __expf(val - mn);
      m = mn;
    }
    sidx = next_sidx;
  }

  // wave-level (m,s) merge; values replicated in 16 lanes -> divide s by 16
#pragma unroll
  for (int off = 16; off >= 1; off >>= 1) {
    const float mo = __shfl_xor(m, off, 32);
    const float so = __shfl_xor(s, off, 32);
    const float mn = fmaxf(m, mo);
    s = s * __expf(m - mn) + so * __expf(mo - mn);
    m = mn;
  }
  s *= (1.0f / 16.0f);
  if (lane == 0) { s_m[wave] = m; s_s[wave] = s; }
  __syncthreads();

  if (wave == 0) {
    // positive score in f32: invnorm[label] * dot(proj[label], x_norm)
    const bool  lab_ok = (label >= 0) && (label < kVocab);
    const float* lrow  = proj + (long)(lab_ok ? label : 0) * kHidden;
    float ps = 0.0f;
#pragma unroll
    for (int i = 0; i < 4; ++i) ps += lrow[lane * 4 + i] * s_x[lane * 4 + i];
    ps = wave_sum32(ps);
    ps *= lab_ok ? invnorm[label] : 0.0f;

    if (lane == 0) {
      float mm = s_m[0], ss = s_s[0];
#pragma unroll
      for (int w = 1; w < kWaves; ++w) {
        const float mo = s_m[w], so = s_s[w];
        const float mn = fmaxf(mm, mo);
        ss = ss * __expf(mm - mn) + so * __expf(mo - mn);
        mm = mn;
      }
      const float mn = fmaxf(mm, ps);              // include positive logit
      ss = ss * __expf(mm - mn) + __expf(ps - mn);
      const float lse     = mn + __logf(ss);
      const float per_tok = lse - ps;
      if (label != kIgnore) {
        atomicAdd(&accum[0], per_tok);
        atomicAdd(&accum[1], 1.0f);
      }
    }
  }
}

// ---------------- kernel 4: finalize ----------------
__global__ void finalize_kernel(const float* __restrict__ accum, float* __restrict__ out) {
  out[0] = accum[0] / fmaxf(accum[1], 1.0f);
}

extern "C" void kernel_launch(void* const* d_in, const int* in_sizes, int n_in,
                              void* d_out, int out_size, void* d_ws, size_t ws_size,
                              hipStream_t stream) {
  const float* inputs  = (const float*)d_in[0];   // [1024,128]
  const int*   labels  = (const int*)d_in[1];     // [1024]
  const int*   samples = (const int*)d_in[2];     // [1024,4096]
  const float* proj    = (const float*)d_in[3];   // [100000,128]
  float* out = (float*)d_out;

  float*    ws      = (float*)d_ws;
  float*    invnorm = ws + kInvnOff;
  float*    xnorm   = ws + kXnormOff;
  float*    accum   = ws + kAccumOff;
  _Float16* projh   = (_Float16*)((char*)d_ws + kProjhByte);

  const bool f16path = (ws_size >= kWsNeeded);   // constant per run -> deterministic

  proj_prep_kernel<<<400, 256, 0, stream>>>(proj, invnorm, projh, kVocab,
                                            f16path ? 1 : 0);
  x_normalize_kernel<<<32, 256, 0, stream>>>(inputs, xnorm, accum);
  if (f16path) {
    sampled_nce_main_kernel<true><<<kTok, 256, 0, stream>>>(
        proj, projh, labels, samples, invnorm, xnorm, accum);
  } else {
    sampled_nce_main_kernel<false><<<kTok, 256, 0, stream>>>(
        proj, projh, labels, samples, invnorm, xnorm, accum);
  }
  finalize_kernel<<<1, 1, 0, stream>>>(accum, out);
}
